// LSTMModel_68436008894971
// MI455X (gfx1250) — compile-verified, matrix operand order checked by reading, not run on hardware
//
#include <hip/hip_runtime.h>
#include <hip/hip_bf16.h>
#include <math.h>

// ---------------------------------------------------------------------------
// Problem constants (from reference)
// ---------------------------------------------------------------------------
#define BATCH   32
#define TENC    128
#define TDEC    64
#define DIM     512          // D == H == 512
#define G4      2048         // 4*H
#define VOCAB   32000

// ---------------------------------------------------------------------------
// Vector types for WMMA
// ---------------------------------------------------------------------------
typedef __attribute__((ext_vector_type(16))) __bf16 v16bf;
typedef __attribute__((ext_vector_type(8)))  __bf16 v8bf;
typedef __attribute__((ext_vector_type(8)))  float  v8f;

__device__ __forceinline__ __bf16 f2bf(float f) {
  unsigned u = __builtin_bit_cast(unsigned, f);
  unsigned r = u + 0x7FFFu + ((u >> 16) & 1u);       // round-to-nearest-even
  unsigned short s = (unsigned short)(r >> 16);
  return __builtin_bit_cast(__bf16, s);
}

// Load one lane's 16 bf16 values of an A/B fragment for 16x16x32 WMMA.
// Per-lane K layout: K = [kb .. kb+7] and [kb+16 .. kb+23] (kb already
// includes the +8 shift for lanes 16..31).  Two aligned 16B loads.
__device__ __forceinline__ v16bf load_frag(const __bf16* __restrict__ p) {
  v8bf lo = *(const v8bf*)(p);
  v8bf hi = *(const v8bf*)(p + 16);
  v16bf r;
#pragma unroll
  for (int i = 0; i < 8; ++i) { r[i] = lo[i]; r[i + 8] = hi[i]; }
  return r;
}

__device__ __forceinline__ v8f wmma_bf16(v16bf a, v16bf b, v8f c) {
  // (neg_a, A, neg_b, B, c_mod, C, reuse_a, reuse_b)
  return __builtin_amdgcn_wmma_f32_16x16x32_bf16(false, a, false, b,
                                                 (short)0, c, false, false);
}

__device__ __forceinline__ float sigmoidf(float x) {
  return 1.0f / (1.0f + __expf(-x));
}

// ---------------------------------------------------------------------------
// Utility kernels
// ---------------------------------------------------------------------------

// zero h(f32), c(f32), out_prev(f32), h_bf16 buffer 0   (all BATCH*DIM)
__global__ void zero_state_kernel(float* h, float* c, float* op, __bf16* hb0) {
  int i = blockIdx.x * blockDim.x + threadIdx.x;
  if (i < BATCH * DIM) {
    h[i] = 0.0f; c[i] = 0.0f; op[i] = 0.0f;
    hb0[i] = f2bf(0.0f);
  }
}

// in: f32 [K,N] row-major  ->  out: bf16 [N,K] row-major (transposed)
__global__ void transpose_bf16_kernel(const float* __restrict__ in,
                                      __bf16* __restrict__ out,
                                      int K, int N) {
  long long idx = (long long)blockIdx.x * blockDim.x + threadIdx.x;
  long long total = (long long)K * N;
  if (idx < total) {
    int n = (int)(idx / K);
    int k = (int)(idx % K);
    out[idx] = f2bf(in[(long long)k * N + n]);
  }
}

// gather embeddings: x[i, d] = bf16(E[tokens[i], d]),  i in [0, NT)
__global__ void gather_embed_kernel(const int* __restrict__ tokens,
                                    const float* __restrict__ E,
                                    __bf16* __restrict__ x, int NT) {
  long long idx = (long long)blockIdx.x * blockDim.x + threadIdx.x;
  long long total = (long long)NT * DIM;
  if (idx < total) {
    int i = (int)(idx / DIM);
    int d = (int)(idx % DIM);
    x[idx] = f2bf(E[(long long)tokens[i] * DIM + d]);
  }
}

// ---------------------------------------------------------------------------
// Generic bf16 WMMA GEMM:  C[M,N] = A[M,K] * Bt[N,K]^T + bias[N]   (f32 out)
// grid: (N/128, M/16), block: 256 threads = 8 waves, one 16x16 tile per wave.
// K multiple of 32.  All operands resident in L2 for our sizes.
// ---------------------------------------------------------------------------
__global__ void gemm_bf16_kernel(const __bf16* __restrict__ A,
                                 const __bf16* __restrict__ Bt,
                                 const float* __restrict__ bias,
                                 float* __restrict__ C,
                                 int N, int K) {
  const int lane = threadIdx.x & 31;
  const int wib  = threadIdx.x >> 5;          // wave in block: 0..7
  const int lrow = lane & 15;
  const int koff = (lane >> 4) * 8;           // +8 K shift for upper half-wave

  const int mt = blockIdx.y;                  // 16-row tile
  const int nt = blockIdx.x * 8 + wib;        // 16-col tile

  const __bf16* ap = A  + (long long)(mt * 16 + lrow) * K + koff;
  const __bf16* bp = Bt + (long long)(nt * 16 + lrow) * K + koff;

  v8f acc = {};
  for (int kk = 0; kk < K; kk += 32) {
    v16bf a = load_frag(ap + kk);
    v16bf b = load_frag(bp + kk);
    acc = wmma_bf16(a, b, acc);
  }

  // C layout: VGPR r -> row m = r + 8*(lane>=16), col n = lane&15
  const int n   = lrow;
  const int col = nt * 16 + n;
  const float bv = bias[col];
#pragma unroll
  for (int r = 0; r < 8; ++r) {
    int m   = r + ((lane >> 4) << 3);
    int row = mt * 16 + m;
    C[(long long)row * N + col] = acc[r] + bv;
  }
}

// ---------------------------------------------------------------------------
// Fused LSTM step:  z = xz[:,t,:] + h @ U ; gates ; masked c/h/out update.
// grid: 8 blocks x 256 threads = 64 waves = 2 (batch tiles) x 32 (H tiles).
// Each wave: 4 gate accumulators, 64 WMMAs over K=512, then elementwise.
// h kept double-buffered in bf16 (read buf, write other buf).
// ---------------------------------------------------------------------------
__global__ void lstm_step_kernel(const __bf16* __restrict__ hb_in,
                                 __bf16* __restrict__ hb_out,
                                 float* __restrict__ h,
                                 float* __restrict__ c,
                                 const float* __restrict__ xz,   // [B,T,4H]
                                 const __bf16* __restrict__ Ut,  // [4H,512]
                                 const int* __restrict__ tokens, // [B,T]
                                 int t, int T, int act_relu,
                                 float* __restrict__ out_prev,   // [B,512] | null
                                 __bf16* __restrict__ yb) {      // [B,T,512]| null
  const int wave = (blockIdx.x * blockDim.x + threadIdx.x) >> 5;  // 0..63
  const int lane = threadIdx.x & 31;
  const int lrow = lane & 15;
  const int lhi  = lane >> 4;
  const int koff = lhi * 8;

  const int mt = wave & 1;        // batch tile (0..1)
  const int nt = wave >> 1;       // H tile (0..31)

  const int arow  = mt * 16 + lrow;          // batch row for A frag
  const int hcolb = nt * 16 + lrow;          // H column for B frags

  const __bf16* ap  = hb_in + arow * DIM + koff;
  const __bf16* bp0 = Ut + (long long)(0 * DIM + hcolb) * DIM + koff;  // i
  const __bf16* bp1 = Ut + (long long)(1 * DIM + hcolb) * DIM + koff;  // f
  const __bf16* bp2 = Ut + (long long)(2 * DIM + hcolb) * DIM + koff;  // g
  const __bf16* bp3 = Ut + (long long)(3 * DIM + hcolb) * DIM + koff;  // o

  v8f acc0 = {}, acc1 = {}, acc2 = {}, acc3 = {};
#pragma unroll 2
  for (int kk = 0; kk < DIM; kk += 32) {
    v16bf a = load_frag(ap + kk);
    acc0 = wmma_bf16(a, load_frag(bp0 + kk), acc0);
    acc1 = wmma_bf16(a, load_frag(bp1 + kk), acc1);
    acc2 = wmma_bf16(a, load_frag(bp2 + kk), acc2);
    acc3 = wmma_bf16(a, load_frag(bp3 + kk), acc3);
  }

  const int n    = lrow;
  const int hcol = nt * 16 + n;
#pragma unroll
  for (int r = 0; r < 8; ++r) {
    int m  = r + (lhi << 3);
    int bb = mt * 16 + m;                       // batch index
    long long xzb = ((long long)bb * T + t) * G4;

    float zi = acc0[r] + xz[xzb + 0 * DIM + hcol];
    float zf = acc1[r] + xz[xzb + 1 * DIM + hcol];
    float zg = acc2[r] + xz[xzb + 2 * DIM + hcol];
    float zo = acc3[r] + xz[xzb + 3 * DIM + hcol];

    float ig = sigmoidf(zi);
    float fg = sigmoidf(zf);
    float og = sigmoidf(zo);
    float gg = act_relu ? fmaxf(zg, 0.0f) : tanhf(zg);

    int   sidx  = bb * DIM + hcol;
    float c_old = c[sidx];
    float h_old = h[sidx];
    float c_new = fg * c_old + ig * gg;
    float h_new = og * (act_relu ? fmaxf(c_new, 0.0f) : tanhf(c_new));

    bool  msk  = tokens[bb * T + t] != 0;
    float c_st = msk ? c_new : c_old;
    float h_st = msk ? h_new : h_old;
    c[sidx] = c_st;
    h[sidx] = h_st;
    hb_out[sidx] = f2bf(h_st);

    if (out_prev) {
      float out = msk ? h_new : out_prev[sidx];
      out_prev[sidx] = out;
      yb[((long long)bb * T + t) * DIM + hcol] = f2bf(out);
    }
  }
}

// ---------------------------------------------------------------------------
// Row softmax in place: one block per row of `N` floats.
// ---------------------------------------------------------------------------
__global__ void softmax_kernel(float* __restrict__ logits, int N) {
  __shared__ float red[256];
  float* p = logits + (long long)blockIdx.x * N;
  const int tid = threadIdx.x;

  float mx = -INFINITY;
  for (int j = tid; j < N; j += 256) mx = fmaxf(mx, p[j]);
  red[tid] = mx; __syncthreads();
  for (int s = 128; s > 0; s >>= 1) {
    if (tid < s) red[tid] = fmaxf(red[tid], red[tid + s]);
    __syncthreads();
  }
  mx = red[0]; __syncthreads();

  float sum = 0.0f;
  for (int j = tid; j < N; j += 256) {
    float e = __expf(p[j] - mx);
    p[j] = e;
    sum += e;
  }
  red[tid] = sum; __syncthreads();
  for (int s = 128; s > 0; s >>= 1) {
    if (tid < s) red[tid] += red[tid + s];
    __syncthreads();
  }
  float inv = 1.0f / red[0];
  __syncthreads();
  for (int j = tid; j < N; j += 256) p[j] *= inv;
}

// ---------------------------------------------------------------------------
// Host launcher
// ---------------------------------------------------------------------------
extern "C" void kernel_launch(void* const* d_in, const int* in_sizes, int n_in,
                              void* d_out, int out_size, void* d_ws, size_t ws_size,
                              hipStream_t stream) {
  (void)in_sizes; (void)n_in; (void)out_size; (void)ws_size;

  const int*   enc_tokens = (const int*)  d_in[0];   // [32,128]
  const int*   dec_tokens = (const int*)  d_in[1];   // [32,64]
  const float* E_enc      = (const float*)d_in[2];   // [32000,512]
  const float* E_dec      = (const float*)d_in[3];
  const float* W_enc      = (const float*)d_in[4];   // [512,2048]
  const float* U_enc      = (const float*)d_in[5];   // [512,2048]
  const float* b_enc      = (const float*)d_in[6];   // [2048]
  const float* W_dec      = (const float*)d_in[7];
  const float* U_dec      = (const float*)d_in[8];
  const float* b_dec      = (const float*)d_in[9];
  const float* W_out      = (const float*)d_in[10];  // [512,32000]
  const float* b_out      = (const float*)d_in[11];  // [32000]

  float* logits = (float*)d_out;                     // [2048,32000]

  // ---- workspace carve ----
  char* w = (char*)d_ws;
  auto carve = [&](size_t bytes) -> void* {
    void* p = (void*)w;
    w += (bytes + 255) & ~(size_t)255;
    return p;
  };
  __bf16* Wt_enc = (__bf16*)carve((size_t)G4 * DIM * 2);       // [2048,512]
  __bf16* Ut_enc = (__bf16*)carve((size_t)G4 * DIM * 2);
  __bf16* Wt_dec = (__bf16*)carve((size_t)G4 * DIM * 2);
  __bf16* Ut_dec = (__bf16*)carve((size_t)G4 * DIM * 2);
  __bf16* Wt_out = (__bf16*)carve((size_t)VOCAB * DIM * 2);    // [32000,512]
  __bf16* x_enc  = (__bf16*)carve((size_t)BATCH * TENC * DIM * 2);
  __bf16* x_dec  = (__bf16*)carve((size_t)BATCH * TDEC * DIM * 2);
  float*  xz_enc = (float*) carve((size_t)BATCH * TENC * G4 * 4);
  float*  xz_dec = (float*) carve((size_t)BATCH * TDEC * G4 * 4);
  float*  h_st   = (float*) carve((size_t)BATCH * DIM * 4);
  float*  c_st   = (float*) carve((size_t)BATCH * DIM * 4);
  float*  op_st  = (float*) carve((size_t)BATCH * DIM * 4);
  __bf16* hb0    = (__bf16*)carve((size_t)BATCH * DIM * 2);    // double buffer
  __bf16* hb1    = (__bf16*)carve((size_t)BATCH * DIM * 2);
  __bf16* y_bf   = (__bf16*)carve((size_t)BATCH * TDEC * DIM * 2);
  __bf16* hbuf[2] = { hb0, hb1 };

  // ---- 1) zero recurrent state ----
  zero_state_kernel<<<(BATCH * DIM + 255) / 256, 256, 0, stream>>>(
      h_st, c_st, op_st, hb0);

  // ---- 2) weight convert + transpose to bf16 [N,K] ----
  {
    long long tw = (long long)DIM * G4;
    int blks = (int)((tw + 255) / 256);
    transpose_bf16_kernel<<<blks, 256, 0, stream>>>(W_enc, Wt_enc, DIM, G4);
    transpose_bf16_kernel<<<blks, 256, 0, stream>>>(U_enc, Ut_enc, DIM, G4);
    transpose_bf16_kernel<<<blks, 256, 0, stream>>>(W_dec, Wt_dec, DIM, G4);
    transpose_bf16_kernel<<<blks, 256, 0, stream>>>(U_dec, Ut_dec, DIM, G4);
    long long to = (long long)DIM * VOCAB;
    transpose_bf16_kernel<<<(int)((to + 255) / 256), 256, 0, stream>>>(
        W_out, Wt_out, DIM, VOCAB);
  }

  // ---- 3) embedding gathers (f32 -> bf16 activations) ----
  {
    long long te = (long long)BATCH * TENC * DIM;
    gather_embed_kernel<<<(int)((te + 255) / 256), 256, 0, stream>>>(
        enc_tokens, E_enc, x_enc, BATCH * TENC);
    long long td = (long long)BATCH * TDEC * DIM;
    gather_embed_kernel<<<(int)((td + 255) / 256), 256, 0, stream>>>(
        dec_tokens, E_dec, x_dec, BATCH * TDEC);
  }

  // ---- 4) input GEMMs: xz = x @ W + b  (WMMA bf16, f32 accum) ----
  gemm_bf16_kernel<<<dim3(G4 / 128, BATCH * TENC / 16), 256, 0, stream>>>(
      x_enc, Wt_enc, b_enc, xz_enc, G4, DIM);
  gemm_bf16_kernel<<<dim3(G4 / 128, BATCH * TDEC / 16), 256, 0, stream>>>(
      x_dec, Wt_dec, b_dec, xz_dec, G4, DIM);

  // ---- 5) encoder recurrence (tanh), 128 fused steps ----
  for (int t = 0; t < TENC; ++t) {
    lstm_step_kernel<<<8, 256, 0, stream>>>(
        hbuf[t & 1], hbuf[(t + 1) & 1], h_st, c_st, xz_enc, Ut_enc,
        enc_tokens, t, TENC, /*act_relu=*/0, nullptr, nullptr);
  }
  // after 128 steps, live h is in hbuf[0]; decoder t=0 reads hbuf[0]. c/h f32 carry over.

  // ---- 6) decoder recurrence (relu), 64 fused steps, emits y (bf16) ----
  for (int t = 0; t < TDEC; ++t) {
    lstm_step_kernel<<<8, 256, 0, stream>>>(
        hbuf[t & 1], hbuf[(t + 1) & 1], h_st, c_st, xz_dec, Ut_dec,
        dec_tokens, t, TDEC, /*act_relu=*/1, op_st, y_bf);
  }

  // ---- 7) output projection: logits = y @ W_out + b_out  (dominant GEMM) ----
  gemm_bf16_kernel<<<dim3(VOCAB / 128, BATCH * TDEC / 16), 256, 0, stream>>>(
      y_bf, Wt_out, b_out, logits, VOCAB, DIM);

  // ---- 8) softmax rows in place on d_out ----
  softmax_kernel<<<BATCH * TDEC, 256, 0, stream>>>(logits, VOCAB);
}
